// TinyFP4Ref_11218454577400
// MI455X (gfx1250) — compile-verified, matrix operand order checked by reading, not run on hardware
//
#include <hip/hip_runtime.h>

typedef _Float16 v16h __attribute__((ext_vector_type(16)));
typedef _Float16 v8h  __attribute__((ext_vector_type(8)));
typedef float    v8f  __attribute__((ext_vector_type(8)));
typedef float    v4f  __attribute__((ext_vector_type(4)));
typedef float    v2f  __attribute__((ext_vector_type(2)));

#define MDIM 32
#define NDIM 8192
#define KDIM 8192
#define NB   (KDIM / 16)     // 512 scale blocks per row
#define KSPLIT 8
#define KPER (KDIM / KSPLIT) // 1024

// --- saturating round-to-nearest-even cast through float8_e4m3fn ---
__device__ __forceinline__ float cast_e4m3(float v) {
    float av = fabsf(v);
    av = fminf(av, 448.0f);
    float r;
    if (av >= 0.015625f) {            // normal range, >= 2^-6
        int e;
        float m = frexpf(av, &e);     // av = m * 2^e, m in [0.5, 1)
        float q = rintf(m * 16.0f);   // RNE onto 3-bit mantissa (+ implicit 1)
        r = ldexpf(q, e - 4);
    } else {                          // subnormals: quantum 2^-9
        r = rintf(av * 512.0f) * 0.001953125f;
    }
    return copysignf(r, v);
}

// --- round to nearest FP4 (e2m1) magnitude; boundary goes DOWN (searchsorted left) ---
__device__ __forceinline__ float round_fp4(float x) {
    float a = fabsf(x);
    float q =
        (a <= 0.25f) ? 0.0f :
        (a <= 0.75f) ? 0.5f :
        (a <= 1.25f) ? 1.0f :
        (a <= 1.75f) ? 1.5f :
        (a <= 2.5f)  ? 2.0f :
        (a <= 3.5f)  ? 3.0f :
        (a <= 5.0f)  ? 4.0f : 6.0f;
    return copysignf(q, x);
}

// ============================================================================
// Kernel 1: dynamic NVFP4 fake-quant of x -> A[m][k] = round_fp4(x/scale)*sf  (f16)
// (global factor input_scale applied at the end; q*sf has <=6 sig bits -> exact f16)
// ============================================================================
__global__ __launch_bounds__(256)
void quant_act_kernel(const float* __restrict__ x,
                      const float* __restrict__ input_scale,
                      _Float16* __restrict__ A) {
    int t = blockIdx.x * blockDim.x + threadIdx.x;   // one thread per (m, block)
    if (t >= MDIM * NB) return;
    int m = t / NB;
    int b = t - m * NB;
    const v4f* px = (const v4f*)(x + (size_t)m * KDIM + b * 16);
    float vals[16];
    float amax = 0.0f;
    #pragma unroll
    for (int i = 0; i < 4; ++i) {
        v4f v = px[i];
        #pragma unroll
        for (int j = 0; j < 4; ++j) {
            vals[4 * i + j] = v[j];
            amax = fmaxf(amax, fabsf(v[j]));
        }
    }
    float is    = input_scale[0];
    float sf    = cast_e4m3(amax / 6.0f / is);
    float scale = sf * is;
    v8h lo, hi;
    if (scale > 0.0f) {
        #pragma unroll
        for (int j = 0; j < 8; ++j) {
            lo[j] = (_Float16)(round_fp4(vals[j]     / scale) * sf);
            hi[j] = (_Float16)(round_fp4(vals[j + 8] / scale) * sf);
        }
    } else {
        lo = (v8h)0; hi = (v8h)0;
    }
    v8h* pa = (v8h*)(A + (size_t)m * KDIM + b * 16);
    pa[0] = lo;
    pa[1] = hi;
}

// ============================================================================
// Kernel 2: split-K WMMA GEMM.  partial[ks][m][n] = sum_{k in split} A[m][k] * (w[n][k]*ws[n][k/16])
// 8 waves/block, each wave: 16 N-rows, two 16x16 C tiles (M=0..15, M=16..31).
// Weight is streamed with nontemporal b128 loads and converted f32->f16 on the fly.
// ============================================================================
__global__ __launch_bounds__(256)
void gemm_wmma_kernel(const float* __restrict__ w,       // [N,K] fp4 values in f32
                      const float* __restrict__ wscale,  // [N, K/16] e4m3 values in f32
                      const _Float16* __restrict__ A,    // [M,K]
                      float* __restrict__ partial) {     // [KSPLIT, M, N]
    const int lane  = threadIdx.x & 31;
    const int wave  = threadIdx.x >> 5;
    const int nl    = lane & 15;      // column within 16-wide N tile / A row (M half)
    const int khalf = lane >> 4;      // 0: K {0..7,16..23}; 1: K {8..15,24..31}
    const int n     = blockIdx.x * 128 + wave * 16 + nl;
    const int ks    = blockIdx.y;
    const int kbeg  = ks * KPER;

    const float*    wrow  = w      + (size_t)n * KDIM;
    const float*    srow  = wscale + (size_t)n * NB;
    const _Float16* a0row = A + (size_t)nl        * KDIM;  // M rows 0..15
    const _Float16* a1row = A + (size_t)(nl + 16) * KDIM;  // M rows 16..31

    v8f c0 = {};   // C tile for M=0..15
    v8f c1 = {};   // C tile for M=16..31

    #pragma unroll 4
    for (int k0 = kbeg; k0 < kbeg + KPER; k0 += 32) {
        const int kb = k0 + khalf * 8;
        // stream weight (single-use 256 MB matrix): nontemporal 128-bit loads
        v4f w0 = __builtin_nontemporal_load((const v4f*)(wrow + kb));
        v4f w1 = __builtin_nontemporal_load((const v4f*)(wrow + kb + 4));
        v4f w2 = __builtin_nontemporal_load((const v4f*)(wrow + kb + 16));
        v4f w3 = __builtin_nontemporal_load((const v4f*)(wrow + kb + 20));
        v2f s  = *(const v2f*)(srow + (k0 >> 4));   // block scales for K blocks b, b+1

        // B fragment (32x16 f16): VGPRs 0..3 <- K kb..kb+7 (scale s0),
        //                         VGPRs 4..7 <- K kb+16..kb+23 (scale s1)
        v16h bv;
        #pragma unroll
        for (int j = 0; j < 4; ++j) {
            bv[j]      = (_Float16)(w0[j] * s[0]);
            bv[4 + j]  = (_Float16)(w1[j] * s[0]);
            bv[8 + j]  = (_Float16)(w2[j] * s[1]);
            bv[12 + j] = (_Float16)(w3[j] * s[1]);
        }

        // A fragments, same K pattern (contiguous f16 16-byte chunks)
        v8h aa0 = *(const v8h*)(a0row + kb);
        v8h aa1 = *(const v8h*)(a0row + kb + 16);
        v16h av0 = __builtin_shufflevector(aa0, aa1, 0,1,2,3,4,5,6,7,8,9,10,11,12,13,14,15);
        v8h ab0 = *(const v8h*)(a1row + kb);
        v8h ab1 = *(const v8h*)(a1row + kb + 16);
        v16h av1 = __builtin_shufflevector(ab0, ab1, 0,1,2,3,4,5,6,7,8,9,10,11,12,13,14,15);

        c0 = __builtin_amdgcn_wmma_f32_16x16x32_f16(false, av0, false, bv,
                                                    (short)0, c0, false, false);
        c1 = __builtin_amdgcn_wmma_f32_16x16x32_f16(false, av1, false, bv,
                                                    (short)0, c1, false, false);
    }

    // C layout: lanes 0..15 -> M = r, lanes 16..31 -> M = 8 + r (N = lane & 15)
    float* pbase = partial + (size_t)ks * MDIM * NDIM;
    const int mbase = (lane < 16) ? 0 : 8;
    #pragma unroll
    for (int r = 0; r < 8; ++r) {
        pbase[(size_t)(mbase + r)      * NDIM + n] = c0[r];
        pbase[(size_t)(mbase + r + 16) * NDIM + n] = c1[r];
    }
}

// ============================================================================
// Kernel 3: reduce split-K partials, apply global scale, add bias
// ============================================================================
__global__ __launch_bounds__(256)
void reduce_bias_kernel(const float* __restrict__ partial,
                        const float* __restrict__ bias,
                        const float* __restrict__ input_scale,
                        const float* __restrict__ wscale2,
                        float* __restrict__ out) {
    int t = blockIdx.x * blockDim.x + threadIdx.x;
    if (t >= MDIM * NDIM) return;
    int n = t & (NDIM - 1);
    float g = input_scale[0] * wscale2[0];
    float acc = 0.0f;
    #pragma unroll
    for (int s = 0; s < KSPLIT; ++s)
        acc += partial[(size_t)s * MDIM * NDIM + t];
    out[t] = acc * g + bias[n];
}

extern "C" void kernel_launch(void* const* d_in, const int* in_sizes, int n_in,
                              void* d_out, int out_size, void* d_ws, size_t ws_size,
                              hipStream_t stream) {
    const float* x        = (const float*)d_in[0];  // [32, 8192]
    const float* w_fp4    = (const float*)d_in[1];  // [8192, 8192]
    const float* bias     = (const float*)d_in[2];  // [8192]
    const float* in_scale = (const float*)d_in[3];  // [1]
    const float* w_scale  = (const float*)d_in[4];  // [8192, 512]
    const float* w_scale2 = (const float*)d_in[5];  // [1]
    float* out = (float*)d_out;                     // [32, 8192]

    // workspace layout: A (f16, 32*8192 = 512 KB) | partials (f32, 8*32*8192 = 8 MB)
    _Float16* A    = (_Float16*)d_ws;
    float* partial = (float*)((char*)d_ws + (size_t)MDIM * KDIM * sizeof(_Float16));

    // 1) activation fake-quant -> f16 A
    quant_act_kernel<<<(MDIM * NB + 255) / 256, 256, 0, stream>>>(x, in_scale, A);

    // 2) split-K WMMA GEMM into partials
    dim3 ggrid(NDIM / 128, KSPLIT);
    gemm_wmma_kernel<<<ggrid, 256, 0, stream>>>(w_fp4, w_scale, A, partial);

    // 3) reduce + global scale + bias
    reduce_bias_kernel<<<(MDIM * NDIM + 255) / 256, 256, 0, stream>>>(
        partial, bias, in_scale, w_scale2, out);
}
